// Attention_23338852286691
// MI455X (gfx1250) — matrix operand order repaired and checked by reading, not compiled, not run on hardware
//
#include <hip/hip_runtime.h>

typedef __attribute__((ext_vector_type(16))) __bf16 v16bf;
typedef __attribute__((ext_vector_type(8)))  __bf16 v8bf;
typedef __attribute__((ext_vector_type(8)))  float  v8f;

#define NSTATE 768
#define NHEAD  12
#define DHEAD  64
#define SEQ    2048
#define BATCH  2
#define MSLOTS 100
#define MPAD   128
#define ROWS   (BATCH*SEQ)     /* 4096 */
#define QKVN   (3*NSTATE)      /* 2304 */
#define MEMN   (2*NSTATE)      /* 1536 */

// ---------------- WMMA fragment helpers (CDNA5 16-bit layouts) ----------------
// A-matrix 16x32: lane = M (0..15), half g: elems 0..7 -> K = g*8 + e, elems 8..15 -> K = 16 + g*8 + (e-8)
__device__ __forceinline__ v16bf load_a_frag(const __bf16* p, int g) {
    v8bf lo = *(const v8bf*)(p + g * 8);
    v8bf hi = *(const v8bf*)(p + 16 + g * 8);
    v16bf r;
#pragma unroll
    for (int i = 0; i < 8; ++i) { r[i] = lo[i]; r[i + 8] = hi[i]; }
    return r;
}
// B-matrix 32x16: lane = N (0..15), half g: elems 0..7 -> K = g*8 + e, elems 8..15 -> K = 16 + g*8 + (e-8)
__device__ __forceinline__ v16bf load_b_frag(const __bf16* p, int g) {
    v8bf lo = *(const v8bf*)(p + g * 8);
    v8bf hi = *(const v8bf*)(p + 16 + g * 8);
    v16bf r;
#pragma unroll
    for (int i = 0; i < 8; ++i) { r[i] = lo[i]; r[i + 8] = hi[i]; }
    return r;
}
__device__ __forceinline__ v8f wmma_bf16(v16bf a, v16bf b, v8f c) {
    return __builtin_amdgcn_wmma_f32_16x16x32_bf16(false, a, false, b, (short)0, c, false, false);
}

// ---------------- conversion kernels ----------------
__global__ void cvt_bf16(const float* __restrict__ s, __bf16* __restrict__ d, int n) {
    int i = blockIdx.x * blockDim.x + threadIdx.x;
    if (i < n) d[i] = (__bf16)s[i];
}
// src[K][N] (row-major) -> dst[N][K] bf16
__global__ void cvt_transpose_bf16(const float* __restrict__ s, __bf16* __restrict__ d, int K, int N) {
    int i = blockIdx.x * blockDim.x + threadIdx.x;
    if (i >= K * N) return;
    int k = i / N, n = i % N;
    d[(size_t)n * K + k] = (__bf16)s[i];
}
__global__ void cvt_pad_rows_bf16(const float* __restrict__ s, __bf16* __restrict__ d,
                                  int rows_src, int rows_dst, int cols) {
    int i = blockIdx.x * blockDim.x + threadIdx.x;
    if (i >= rows_dst * cols) return;
    int r = i / cols;
    d[i] = (r < rows_src) ? (__bf16)s[i] : (__bf16)0.0f;
}
// v_t[b][h][d][s] from qkv[b*S+s][1536 + h*64 + d]
__global__ void build_vt(const __bf16* __restrict__ qkv, __bf16* __restrict__ vt) {
    int i = blockIdx.x * blockDim.x + threadIdx.x;
    if (i >= BATCH * NHEAD * DHEAD * SEQ) return;
    int s = i % SEQ; int t = i / SEQ;
    int d = t % DHEAD; t /= DHEAD;
    int h = t % NHEAD; int b = t / NHEAD;
    vt[i] = qkv[(size_t)(b * SEQ + s) * QKVN + 2 * NSTATE + h * DHEAD + d];
}
// mv_t[h][d][m] from mkv[m][768 + h*64 + d]
__global__ void build_mvt(const __bf16* __restrict__ mkv, __bf16* __restrict__ mvt) {
    int i = blockIdx.x * blockDim.x + threadIdx.x;
    if (i >= NHEAD * DHEAD * MPAD) return;
    int m = i % MPAD; int t = i / MPAD;
    int d = t % DHEAD; int h = t / DHEAD;
    mvt[i] = mkv[(size_t)m * MEMN + NSTATE + h * DHEAD + d];
}

// ---------------- generic NT GEMM: C[M,N] = A[M,K](bf16) * Bt[N,K](bf16)^T + bias ----------------
__global__ __launch_bounds__(32)
void wmma_gemm_nt(const __bf16* __restrict__ A, const __bf16* __restrict__ Bt,
                  const float* __restrict__ bias,
                  float* __restrict__ Cf, __bf16* __restrict__ Cb, int N, int K) {
    const int m0 = blockIdx.x * 16, n0 = blockIdx.y * 64;
    const int lane = threadIdx.x, lr = lane & 15, g = lane >> 4;
    v8f a0 = {}, a1 = {}, a2 = {}, a3 = {};
    const __bf16* ar = A + (size_t)(m0 + lr) * K;
    const __bf16* b0 = Bt + (size_t)(n0 + lr) * K;
    const __bf16* b1 = Bt + (size_t)(n0 + 16 + lr) * K;
    const __bf16* b2 = Bt + (size_t)(n0 + 32 + lr) * K;
    const __bf16* b3 = Bt + (size_t)(n0 + 48 + lr) * K;
    for (int k0 = 0; k0 < K; k0 += 32) {
        v16bf af = load_a_frag(ar + k0, g);
        a0 = wmma_bf16(af, load_b_frag(b0 + k0, g), a0);
        a1 = wmma_bf16(af, load_b_frag(b1 + k0, g), a1);
        a2 = wmma_bf16(af, load_b_frag(b2 + k0, g), a2);
        a3 = wmma_bf16(af, load_b_frag(b3 + k0, g), a3);
    }
    v8f acc[4] = {a0, a1, a2, a3};
#pragma unroll
    for (int t = 0; t < 4; ++t) {
        int n = n0 + t * 16 + lr;
        float bv = bias ? bias[n] : 0.0f;
#pragma unroll
        for (int r = 0; r < 8; ++r) {
            int m = m0 + g * 8 + r;
            float v = acc[t][r] + bv;
            if (Cf) Cf[(size_t)m * N + n] = v;
            if (Cb) Cb[(size_t)m * N + n] = (__bf16)v;
        }
    }
}

// ---------------- flash self-attention: one wave per (b,h,16 q-rows) ----------------
__global__ __launch_bounds__(32)
void attn_self(const __bf16* __restrict__ qkv, const __bf16* __restrict__ vt,
               float* __restrict__ a_f32, __bf16* __restrict__ cat) {
    __shared__ __bf16 pt[16 * 32];
    const int qt = blockIdx.x, h = blockIdx.y, b = blockIdx.z;
    const int lane = threadIdx.x, lr = lane & 15, g = lane >> 4;
    const int m0 = qt * 16;
    const __bf16* qrow = qkv + (size_t)(b * SEQ + m0 + lr) * QKVN + h * DHEAD;
    v16bf qf0 = load_a_frag(qrow, g);
    v16bf qf1 = load_a_frag(qrow + 32, g);
    const __bf16* kb = qkv + (size_t)b * SEQ * QKVN + NSTATE + h * DHEAD;
    const __bf16* vb = vt + (size_t)(b * NHEAD + h) * DHEAD * SEQ;
    float mi[8], li[8];
    v8f o0 = {}, o1 = {}, o2 = {}, o3 = {};
#pragma unroll
    for (int r = 0; r < 8; ++r) { mi[r] = -__builtin_inff(); li[r] = 0.0f; }
    for (int s0 = 0; s0 < SEQ; s0 += 32) {
        const __bf16* k0p = kb + (size_t)(s0 + lr) * QKVN;
        const __bf16* k1p = kb + (size_t)(s0 + 16 + lr) * QKVN;
        v8f S0 = {}, S1 = {};
        S0 = wmma_bf16(qf0, load_b_frag(k0p, g), S0);
        S0 = wmma_bf16(qf1, load_b_frag(k0p + 32, g), S0);
        S1 = wmma_bf16(qf0, load_b_frag(k1p, g), S1);
        S1 = wmma_bf16(qf1, load_b_frag(k1p + 32, g), S1);
#pragma unroll
        for (int r = 0; r < 8; ++r) {  // row stats: reduce across the 16-lane half
            float mn = fmaxf(S0[r], S1[r]);
            mn = fmaxf(mn, __shfl_xor(mn, 1));
            mn = fmaxf(mn, __shfl_xor(mn, 2));
            mn = fmaxf(mn, __shfl_xor(mn, 4));
            mn = fmaxf(mn, __shfl_xor(mn, 8));
            mn = fmaxf(mn, mi[r]);
            float sc = __expf(mi[r] - mn);
            float p0 = __expf(S0[r] - mn);
            float p1 = __expf(S1[r] - mn);
            float rs = p0 + p1;
            rs += __shfl_xor(rs, 1);
            rs += __shfl_xor(rs, 2);
            rs += __shfl_xor(rs, 4);
            rs += __shfl_xor(rs, 8);
            li[r] = li[r] * sc + rs;
            mi[r] = mn;
            o0[r] *= sc; o1[r] *= sc; o2[r] *= sc; o3[r] *= sc;
            int mr = g * 8 + r;  // C-layout -> row-major P in LDS
            pt[mr * 32 + lr] = (__bf16)p0;
            pt[mr * 32 + 16 + lr] = (__bf16)p1;
        }
        asm volatile("s_wait_dscnt 0" ::: "memory");
        v16bf pf = load_a_frag(&pt[lr * 32], g);
        o0 = wmma_bf16(pf, load_b_frag(vb + (size_t)(lr) * SEQ + s0, g), o0);
        o1 = wmma_bf16(pf, load_b_frag(vb + (size_t)(16 + lr) * SEQ + s0, g), o1);
        o2 = wmma_bf16(pf, load_b_frag(vb + (size_t)(32 + lr) * SEQ + s0, g), o2);
        o3 = wmma_bf16(pf, load_b_frag(vb + (size_t)(48 + lr) * SEQ + s0, g), o3);
    }
    v8f oo[4] = {o0, o1, o2, o3};
#pragma unroll
    for (int t = 0; t < 4; ++t) {
        int d = t * 16 + lr;
#pragma unroll
        for (int r = 0; r < 8; ++r) {
            size_t row = (size_t)(b * SEQ + m0 + g * 8 + r);
            float v = oo[t][r] / li[r];
            a_f32[row * NSTATE + h * DHEAD + d] = v;
            cat[row * MEMN + h * DHEAD + d] = (__bf16)v;
        }
    }
}

// ---------------- memory attention: logits 16x128 in registers ----------------
__global__ __launch_bounds__(32)
void attn_mem(const __bf16* __restrict__ qkv, const __bf16* __restrict__ mkv,
              const __bf16* __restrict__ mvt,
              float* __restrict__ a1_f32, __bf16* __restrict__ cat) {
    __shared__ __bf16 pt[16 * MPAD];
    const int qt = blockIdx.x, h = blockIdx.y, b = blockIdx.z;
    const int lane = threadIdx.x, lr = lane & 15, g = lane >> 4;
    const int m0 = qt * 16;
    const __bf16* qrow = qkv + (size_t)(b * SEQ + m0 + lr) * QKVN + h * DHEAD;
    v16bf qf0 = load_a_frag(qrow, g);
    v16bf qf1 = load_a_frag(qrow + 32, g);
    v8f S[8];
#pragma unroll
    for (int j = 0; j < 8; ++j) {
        const __bf16* kp = mkv + (size_t)(j * 16 + lr) * MEMN + h * DHEAD;
        v8f s = {};
        s = wmma_bf16(qf0, load_b_frag(kp, g), s);
        s = wmma_bf16(qf1, load_b_frag(kp + 32, g), s);
        S[j] = s;
    }
    float li[8];
#pragma unroll
    for (int r = 0; r < 8; ++r) {
        float mx = -__builtin_inff();
#pragma unroll
        for (int j = 0; j < 8; ++j) {
            if (j * 16 + lr >= MSLOTS) S[j][r] = -__builtin_inff();  // mask pad slots
            mx = fmaxf(mx, S[j][r]);
        }
        mx = fmaxf(mx, __shfl_xor(mx, 1));
        mx = fmaxf(mx, __shfl_xor(mx, 2));
        mx = fmaxf(mx, __shfl_xor(mx, 4));
        mx = fmaxf(mx, __shfl_xor(mx, 8));
        float rs = 0.0f;
#pragma unroll
        for (int j = 0; j < 8; ++j) { float p = __expf(S[j][r] - mx); S[j][r] = p; rs += p; }
        rs += __shfl_xor(rs, 1);
        rs += __shfl_xor(rs, 2);
        rs += __shfl_xor(rs, 4);
        rs += __shfl_xor(rs, 8);
        li[r] = rs;
        int mr = g * 8 + r;
#pragma unroll
        for (int j = 0; j < 8; ++j) pt[mr * MPAD + j * 16 + lr] = (__bf16)S[j][r];
    }
    asm volatile("s_wait_dscnt 0" ::: "memory");
    v8f o0 = {}, o1 = {}, o2 = {}, o3 = {};
#pragma unroll
    for (int c = 0; c < 4; ++c) {
        v16bf pf = load_a_frag(&pt[lr * MPAD + c * 32], g);
        o0 = wmma_bf16(pf, load_b_frag(mvt + (size_t)(h * DHEAD + lr) * MPAD + c * 32, g), o0);
        o1 = wmma_bf16(pf, load_b_frag(mvt + (size_t)(h * DHEAD + 16 + lr) * MPAD + c * 32, g), o1);
        o2 = wmma_bf16(pf, load_b_frag(mvt + (size_t)(h * DHEAD + 32 + lr) * MPAD + c * 32, g), o2);
        o3 = wmma_bf16(pf, load_b_frag(mvt + (size_t)(h * DHEAD + 48 + lr) * MPAD + c * 32, g), o3);
    }
    v8f oo[4] = {o0, o1, o2, o3};
#pragma unroll
    for (int t = 0; t < 4; ++t) {
        int d = t * 16 + lr;
#pragma unroll
        for (int r = 0; r < 8; ++r) {
            size_t row = (size_t)(b * SEQ + m0 + g * 8 + r);
            float v = oo[t][r] / li[r];
            a1_f32[row * NSTATE + h * DHEAD + d] = v;
            cat[row * MEMN + NSTATE + h * DHEAD + d] = (__bf16)v;
        }
    }
}

// ---------------- alpha gate + blend: sigmoid(cat @ w_alpha + b) epilogue ----------------
__global__ __launch_bounds__(32)
void alpha_fuse(const __bf16* __restrict__ cat, const __bf16* __restrict__ wat,
                const float* __restrict__ b_alpha, const float* __restrict__ a,
                const float* __restrict__ a1, __bf16* __restrict__ fused) {
    const int m0 = blockIdx.x * 16, n0 = blockIdx.y * 64;
    const int lane = threadIdx.x, lr = lane & 15, g = lane >> 4;
    v8f c0 = {}, c1 = {}, c2 = {}, c3 = {};
    const __bf16* ar = cat + (size_t)(m0 + lr) * MEMN;
    const __bf16* b0 = wat + (size_t)(n0 + lr) * MEMN;
    const __bf16* b1 = wat + (size_t)(n0 + 16 + lr) * MEMN;
    const __bf16* b2 = wat + (size_t)(n0 + 32 + lr) * MEMN;
    const __bf16* b3 = wat + (size_t)(n0 + 48 + lr) * MEMN;
    for (int k0 = 0; k0 < MEMN; k0 += 32) {
        v16bf af = load_a_frag(ar + k0, g);
        c0 = wmma_bf16(af, load_b_frag(b0 + k0, g), c0);
        c1 = wmma_bf16(af, load_b_frag(b1 + k0, g), c1);
        c2 = wmma_bf16(af, load_b_frag(b2 + k0, g), c2);
        c3 = wmma_bf16(af, load_b_frag(b3 + k0, g), c3);
    }
    v8f acc[4] = {c0, c1, c2, c3};
#pragma unroll
    for (int t = 0; t < 4; ++t) {
        int n = n0 + t * 16 + lr;
        float bv = b_alpha[n];
#pragma unroll
        for (int r = 0; r < 8; ++r) {
            size_t m = (size_t)(m0 + g * 8 + r);
            float al = 1.0f / (1.0f + __expf(-(acc[t][r] + bv)));
            float av = a[m * NSTATE + n], a1v = a1[m * NSTATE + n];
            fused[m * NSTATE + n] = (__bf16)(al * av + (1.0f - al) * a1v);
        }
    }
}

// ---------------- host orchestration ----------------
extern "C" void kernel_launch(void* const* d_in, const int* in_sizes, int n_in,
                              void* d_out, int out_size, void* d_ws, size_t ws_size,
                              hipStream_t stream) {
    (void)in_sizes; (void)n_in; (void)out_size; (void)ws_size;
    const float* x       = (const float*)d_in[0];
    const float* w_attn  = (const float*)d_in[1];
    const float* b_attn  = (const float*)d_in[2];
    const float* w_proj  = (const float*)d_in[3];
    const float* b_proj  = (const float*)d_in[4];
    const float* w_mem   = (const float*)d_in[5];
    const float* b_mem   = (const float*)d_in[6];
    const float* w_alpha = (const float*)d_in[7];
    const float* b_alpha = (const float*)d_in[8];
    const float* memf    = (const float*)d_in[9];
    float* out = (float*)d_out;

    char* p = (char*)d_ws;
    auto alloc = [&](size_t bytes) { char* r = p; p += (bytes + 255) & ~(size_t)255; return r; };
    __bf16* x_bf   = (__bf16*)alloc((size_t)ROWS * NSTATE * 2);
    __bf16* wat_t  = (__bf16*)alloc((size_t)QKVN * NSTATE * 2);   // w_attn^T [2304][768]
    __bf16* wm_t   = (__bf16*)alloc((size_t)MEMN * NSTATE * 2);   // w_mem^T  [1536][768]
    __bf16* wal_t  = (__bf16*)alloc((size_t)NSTATE * MEMN * 2);   // w_alpha^T[768][1536]
    __bf16* wp_t   = (__bf16*)alloc((size_t)NSTATE * NSTATE * 2); // w_proj^T [768][768]
    __bf16* mem_bf = (__bf16*)alloc((size_t)MPAD * NSTATE * 2);
    __bf16* qkv_bf = (__bf16*)alloc((size_t)ROWS * QKVN * 2);
    __bf16* mkv_bf = (__bf16*)alloc((size_t)MPAD * MEMN * 2);
    __bf16* vt     = (__bf16*)alloc((size_t)BATCH * NHEAD * DHEAD * SEQ * 2);
    __bf16* mvt    = (__bf16*)alloc((size_t)NHEAD * DHEAD * MPAD * 2);
    float*  a_f    = (float*)alloc((size_t)ROWS * NSTATE * 4);
    float*  a1_f   = (float*)alloc((size_t)ROWS * NSTATE * 4);
    __bf16* cat_bf = (__bf16*)alloc((size_t)ROWS * MEMN * 2);
    __bf16* fus_bf = (__bf16*)alloc((size_t)ROWS * NSTATE * 2);

    auto g1 = [](int n) { return dim3((n + 255) / 256); };
    cvt_bf16<<<g1(ROWS * NSTATE), 256, 0, stream>>>(x, x_bf, ROWS * NSTATE);
    cvt_transpose_bf16<<<g1(NSTATE * QKVN), 256, 0, stream>>>(w_attn, wat_t, NSTATE, QKVN);
    cvt_transpose_bf16<<<g1(NSTATE * MEMN), 256, 0, stream>>>(w_mem, wm_t, NSTATE, MEMN);
    cvt_transpose_bf16<<<g1(MEMN * NSTATE), 256, 0, stream>>>(w_alpha, wal_t, MEMN, NSTATE);
    cvt_transpose_bf16<<<g1(NSTATE * NSTATE), 256, 0, stream>>>(w_proj, wp_t, NSTATE, NSTATE);
    cvt_pad_rows_bf16<<<g1(MPAD * NSTATE), 256, 0, stream>>>(memf, mem_bf, MSLOTS, MPAD, NSTATE);

    // qkv = x @ w_attn + b_attn   [4096 x 2304]
    wmma_gemm_nt<<<dim3(ROWS / 16, QKVN / 64), 32, 0, stream>>>(x_bf, wat_t, b_attn, nullptr, qkv_bf, QKVN, NSTATE);
    // mkv = mem_pad @ w_mem + b_mem  [128 x 1536]
    wmma_gemm_nt<<<dim3(MPAD / 16, MEMN / 64), 32, 0, stream>>>(mem_bf, wm_t, b_mem, nullptr, mkv_bf, MEMN, NSTATE);

    build_vt<<<g1(BATCH * NHEAD * DHEAD * SEQ), 256, 0, stream>>>(qkv_bf, vt);
    build_mvt<<<g1(NHEAD * DHEAD * MPAD), 256, 0, stream>>>(mkv_bf, mvt);

    attn_self<<<dim3(SEQ / 16, NHEAD, BATCH), 32, 0, stream>>>(qkv_bf, vt, a_f, cat_bf);
    attn_mem<<<dim3(SEQ / 16, NHEAD, BATCH), 32, 0, stream>>>(qkv_bf, mkv_bf, mvt, a1_f, cat_bf);

    alpha_fuse<<<dim3(ROWS / 16, NSTATE / 64), 32, 0, stream>>>(cat_bf, wal_t, b_alpha, a_f, a1_f, fus_bf);

    // out = fused @ w_proj + b_proj  [4096 x 768] fp32
    wmma_gemm_nt<<<dim3(ROWS / 16, NSTATE / 64), 32, 0, stream>>>(fus_bf, wp_t, b_proj, out, nullptr, NSTATE, NSTATE);
}
